// block_wise_choice_57475252355408
// MI455X (gfx1250) — matrix operand order repaired and checked by reading, not compiled, not run on hardware
//
#include <hip/hip_runtime.h>
#include <hip/hip_bf16.h>
#include <stdint.h>

typedef __attribute__((ext_vector_type(16))) __bf16 v16bf;
typedef __attribute__((ext_vector_type(8)))  float  v8f;
typedef __attribute__((ext_vector_type(4)))  unsigned u32x4;
typedef __attribute__((ext_vector_type(8)))  int      i32x8;
typedef __attribute__((ext_vector_type(4)))  int      i32x4;

union FragBF { v16bf v; uint4 u[2]; };

#define BB   4
#define TT   2048
#define CC   1024
#define HH   4
#define DD   256
#define QD   1024
#define BLK  64
#define COLK 1024

// gfx1250 async global->LDS copy (ASYNCcnt tracked), GVS addressing:
// mem = SADDR(base) + VADDR(byte offset); VDST = wave-relative LDS byte offset.
__device__ __forceinline__ void async_load_b128(unsigned ldsOff, unsigned gOff,
                                                const void* base)
{
  asm volatile("global_load_async_to_lds_b128 %0, %1, %2"
               :: "v"(ldsOff), "v"(gOff), "s"(base) : "memory");
}

// gfx1250 Tensor Data Mover: one instruction moves a 2D tile (tileW x tileH
// elements of 2 bytes, row stride rowStrideElems) from global to LDS, adding
// pad_amount DWORDs of LDS padding every 2^(padIv+1) DWORDs (keeps our padded
// LDS row pitch / bank-conflict avoidance for free).  TENSORcnt tracked.
__device__ __forceinline__ void tdm_load_2d(unsigned ldsOff, const void* gaddr,
                                            unsigned tileW, unsigned tileH,
                                            unsigned rowStrideElems,
                                            unsigned padIv, unsigned padAmt)
{
  unsigned long long ga = (unsigned long long)gaddr;
  u32x4 g0;
  g0.x = 1u;                                  // count=1 (valid user descriptor)
  g0.y = ldsOff;                              // LDS byte address
  g0.z = (unsigned)ga;                        // global addr [31:0]
  g0.w = (unsigned)(ga >> 32) | (2u << 30);   // global addr [56:32] | type=2
  i32x8 g1;
  g1[0] = (1 << 16)                           // data_size = 2 bytes
        | (1 << 20)                           // pad_enable
        | ((int)padIv  << 22)                 // pad every 2^(padIv+1) DWORDs
        | ((int)padAmt << 25);                // pad (padAmt+1) DWORDs
  g1[1] = 0;                                  // atomic_barrier=0 | tensor_dim0 lo16 (big dims below)
  g1[2] = (int)(0x4000u | 0u);                // tensor_dim0 hi16 (=2^30) | tensor_dim1 lo16
  g1[3] = (int)(0x4000u | (tileW << 16));     // tensor_dim1 hi16 (=2^30) | tile_dim0
  g1[4] = (int)tileH;                         // tile_dim1 | tile_dim2=0
  g1[5] = (int)rowStrideElems;                // tensor_dim0_stride lo32
  g1[6] = 0;                                  // stride hi16 | dim1_stride lo16
  g1[7] = 0;
  i32x4 z4 = {0, 0, 0, 0};
#if __clang_major__ >= 23
  i32x8 z8 = {0, 0, 0, 0, 0, 0, 0, 0};
  __builtin_amdgcn_tensor_load_to_lds(g0, g1, z4, z4, z8, 0);
#else
  __builtin_amdgcn_tensor_load_to_lds(g0, g1, z4, z4, 0);
#endif
}

// ---------------------------------------------------------------------------
// Kernel 1: causal depthwise conv1d (K=3, left pad 2) + bf16 pack of y
// ---------------------------------------------------------------------------
__global__ __launch_bounds__(256) void conv_pack_kernel(
    const float* __restrict__ x, const float* __restrict__ cw,
    const float* __restrict__ cb, __bf16* __restrict__ Y)
{
  size_t i = (size_t)blockIdx.x * 256 + threadIdx.x;   // over B*T*C
  int c = (int)(i & (CC - 1));
  size_t bt = i >> 10;
  int t = (int)(bt & (TT - 1));
  size_t b = bt >> 11;
  float acc = cb[c];
  #pragma unroll
  for (int j = 0; j < 3; ++j) {
    int tt = t + j - 2;
    if (tt >= 0)
      acc += x[(b * TT + tt) * CC + c] * cw[c * 3 + j];
  }
  Y[i] = (__bf16)acc;
}

__global__ __launch_bounds__(256) void cvt_bf16_kernel(
    const float* __restrict__ in, __bf16* __restrict__ out, int n)
{
  int i = blockIdx.x * 256 + threadIdx.x;
  if (i < n) out[i] = (__bf16)in[i];
}

// ---------------------------------------------------------------------------
// Kernel 2: q = y @ lin_w^T + lin_b   (M=8192, N=1024, K=1024), bf16 WMMA
// 256 thr (8 waves), tile 128M x 128N, K stepped by 32.
// Double-buffered LDS fed by global_load_async_to_lds_b128; ASYNCcnt in-order
// completion lets "s_wait_asynccnt 4" wait on the previous tile only while the
// next tile is still in flight.
// ---------------------------------------------------------------------------
__global__ __launch_bounds__(256) void gemm_q_wmma(
    const __bf16* __restrict__ Y, const __bf16* __restrict__ W,
    const float* __restrict__ lin_b, __bf16* __restrict__ Q)
{
  constexpr int BM = 128, BN = 128, BK = 32, LDP = 40;  // padded LDS row (halves)
  __shared__ __bf16 As[2][BM * LDP];
  __shared__ __bf16 Bs[2][BN * LDP];
  const int tid  = threadIdx.x;
  const int lane = tid & 31;
  const int wv   = tid >> 5;        // 0..7, owns rows [wv*16, wv*16+16)
  const int g    = lane >> 4;       // half-wave group
  const int nl   = lane & 15;
  const int m0 = blockIdx.y * BM;
  const int n0 = blockIdx.x * BN;

  // per-thread staging slots (two 8-half chunks each for A and B)
  int row0 = tid >> 2,            c80 = (tid & 3) * 8;
  int row1 = (tid + 256) >> 2,    c81 = ((tid + 256) & 3) * 8;

  v8f acc[8];
  #pragma unroll
  for (int s = 0; s < 8; ++s) {
    float bias = lin_b[n0 + s * 16 + nl];
    #pragma unroll
    for (int r = 0; r < 8; ++r) acc[s][r] = bias;
  }

  auto issue_tile = [&](int buf, int k0) {
    unsigned ldsA0 = (unsigned)(size_t)(&As[buf][row0 * LDP + c80]);
    unsigned ldsB0 = (unsigned)(size_t)(&Bs[buf][row0 * LDP + c80]);
    unsigned ldsA1 = (unsigned)(size_t)(&As[buf][row1 * LDP + c81]);
    unsigned ldsB1 = (unsigned)(size_t)(&Bs[buf][row1 * LDP + c81]);
    async_load_b128(ldsA0, (unsigned)(((m0 + row0) * CC + k0 + c80) * 2), Y);
    async_load_b128(ldsB0, (unsigned)(((n0 + row0) * CC + k0 + c80) * 2), W);
    async_load_b128(ldsA1, (unsigned)(((m0 + row1) * CC + k0 + c81) * 2), Y);
    async_load_b128(ldsB1, (unsigned)(((n0 + row1) * CC + k0 + c81) * 2), W);
  };

  issue_tile(0, 0);
  int cur = 0;
  for (int k0 = 0; k0 < CC; k0 += BK) {
    if (k0 + BK < CC) {
      issue_tile(cur ^ 1, k0 + BK);                       // prefetch next tile
      asm volatile("s_wait_asynccnt 0x4" ::: "memory");   // previous tile done
    } else {
      asm volatile("s_wait_asynccnt 0x0" ::: "memory");
    }
    __syncthreads();

    // A fragment (16x32): lanes 0-15 K=0..15, lanes 16-31 K=16..31
    FragBF fa, fb[8];
    const __bf16* arow = &As[cur][(wv * 16 + nl) * LDP];
    fa.u[0] = *(const uint4*)(arow + 16 * g);
    fa.u[1] = *(const uint4*)(arow + 16 * g + 8);
    #pragma unroll
    for (int s = 0; s < 8; ++s) {
      // B fragment (32x16): lane group g holds K = 16g + j, N = nl
      const __bf16* brow = &Bs[cur][(s * 16 + nl) * LDP];
      fb[s].u[0] = *(const uint4*)(brow + 16 * g);
      fb[s].u[1] = *(const uint4*)(brow + 16 * g + 8);
    }
    #pragma unroll
    for (int s = 0; s < 8; ++s)
      acc[s] = __builtin_amdgcn_wmma_f32_16x16x32_bf16(
          false, fa.v, false, fb[s].v, (short)0, acc[s], false, false);

    __syncthreads();
    cur ^= 1;
  }

  // D layout: lane nl = N, vgpr r -> M = r + 8*g
  #pragma unroll
  for (int s = 0; s < 8; ++s) {
    #pragma unroll
    for (int r = 0; r < 8; ++r) {
      int m = m0 + wv * 16 + r + 8 * g;
      int n = n0 + s * 16 + nl;
      Q[(size_t)m * QD + n] = (__bf16)acc[s][r];
    }
  }
}

// ---------------------------------------------------------------------------
// Kernel 3: per (b,h) fused col-score GEMM + rowwise max/argmax + row-score
// GEMM + score write.  All global->LDS staging is done by the Tensor Data
// Mover: wave 0 issues one tensor_load_to_lds per tile and runs a TENSORcnt
// double-buffer pipeline (issue tile s+1, wait <=1 => tile s resident).
// 128 thr (4 waves), tile 64M; A cached in LDS for the whole K=256.
// ---------------------------------------------------------------------------
__global__ __launch_bounds__(128) void scores_wmma(
    const __bf16* __restrict__ Q, const __bf16* __restrict__ CK,
    const __bf16* __restrict__ RK, float* __restrict__ scoreOut,
    int* __restrict__ idxOut)
{
  constexpr int BMt = 64, BKt = 32, LDA = 264, LDB = 40;
  __shared__ __bf16 As[BMt * LDA];     // 64 x 256 (+16B/row TDM pad)
  __shared__ __bf16 Bs[2][128 * LDB];  // double-buffered key tiles
  const int tid  = threadIdx.x;
  const int lane = tid & 31;
  const int wv   = tid >> 5;           // 0..3
  const int g    = lane >> 4;
  const int nl   = lane & 15;
  const int bh   = blockIdx.y;         // b*H + h
  const int h    = bh & 3;
  const int m0   = blockIdx.x * BMt;

  const __bf16* Abase = Q + (size_t)bh * (TT * DD) + (size_t)m0 * DD;

  // TDM pipeline prologue (wave 0 only; EXEC-ignored scalar-side DMA):
  //  A tile: 256x64, contiguous rows (512B), pad 4 DWORDs every 128 DWORDs.
  //  First colkey tile: 32x128, row stride H*D elems, pad 4 every 16 DWORDs.
  if (wv == 0) {
    tdm_load_2d((unsigned)(size_t)As, Abase, DD, BMt, DD, 6, 3);
    tdm_load_2d((unsigned)(size_t)&Bs[0][0], CK + (size_t)h * DD,
                BKt, 128, HH * DD, 3, 3);
  }

  float rmax[8]; int rarg[8];
  #pragma unroll
  for (int r = 0; r < 8; ++r) { rmax[r] = -3.0e38f; rarg[r] = 0; }

  // ---- column keys: 8 N-tiles of 128, fused running max/argmax ----
  for (int nt = 0; nt < 8; ++nt) {
    v8f acc[8];
    #pragma unroll
    for (int s = 0; s < 8; ++s)
      #pragma unroll
      for (int r = 0; r < 8; ++r) acc[s][r] = 0.0f;

    for (int ks = 0; ks < 8; ++ks) {
      const int s = nt * 8 + ks;
      if (wv == 0) {
        const int s1 = s + 1;
        if (s1 < 64) {        // next colkey tile
          const int nt1 = s1 >> 3, ks1 = s1 & 7;
          tdm_load_2d((unsigned)(size_t)&Bs[s1 & 1][0],
                      CK + ((size_t)(nt1 * 128) * HH + h) * DD + ks1 * BKt,
                      BKt, 128, HH * DD, 3, 3);
        } else {              // pipeline straight into first rowkey tile
          tdm_load_2d((unsigned)(size_t)&Bs[s1 & 1][0],
                      RK + (size_t)h * DD, BKt, BLK, HH * DD, 3, 3);
        }
        __builtin_amdgcn_s_wait_tensorcnt((short)1);   // tile s resident
      }
      __syncthreads();

      FragBF fa, fb[8];
      const __bf16* arow = As + (wv * 16 + nl) * LDA + ks * BKt;
      fa.u[0] = *(const uint4*)(arow + 16 * g);
      fa.u[1] = *(const uint4*)(arow + 16 * g + 8);
      const __bf16* bbuf = &Bs[s & 1][0];
      #pragma unroll
      for (int ss = 0; ss < 8; ++ss) {
        const __bf16* brow = bbuf + (ss * 16 + nl) * LDB;
        fb[ss].u[0] = *(const uint4*)(brow + 16 * g);
        fb[ss].u[1] = *(const uint4*)(brow + 16 * g + 8);
      }
      #pragma unroll
      for (int ss = 0; ss < 8; ++ss)
        acc[ss] = __builtin_amdgcn_wmma_f32_16x16x32_bf16(
            false, fa.v, false, fb[ss].v, (short)0, acc[ss], false, false);
      __syncthreads();
    }
    #pragma unroll
    for (int s = 0; s < 8; ++s) {
      int n = nt * 128 + s * 16 + nl;
      #pragma unroll
      for (int r = 0; r < 8; ++r) {
        float v = acc[s][r];
        if (v > rmax[r]) { rmax[r] = v; rarg[r] = n; }  // strict > keeps first idx
      }
    }
  }

  // cross-lane max/argmax inside each 16-lane half (same M row, tie -> low idx)
  #pragma unroll
  for (int r = 0; r < 8; ++r) {
    #pragma unroll
    for (int mk = 1; mk <= 8; mk <<= 1) {
      float ov = __shfl_xor(rmax[r], mk, 32);
      int   oa = __shfl_xor(rarg[r], mk, 32);
      if (ov > rmax[r] || (ov == rmax[r] && oa < rarg[r])) { rmax[r] = ov; rarg[r] = oa; }
    }
  }

  // ---- row keys: 64 columns = 4 WMMA subtiles (TDM pipeline continues) ----
  v8f accR[4];
  #pragma unroll
  for (int s = 0; s < 4; ++s)
    #pragma unroll
    for (int r = 0; r < 8; ++r) accR[s][r] = 0.0f;

  for (int ks = 0; ks < 8; ++ks) {
    const int s = 64 + ks;
    if (wv == 0) {
      if (ks < 7)
        tdm_load_2d((unsigned)(size_t)&Bs[(s + 1) & 1][0],
                    RK + (size_t)h * DD + (ks + 1) * BKt,
                    BKt, BLK, HH * DD, 3, 3);
      if (ks < 7) __builtin_amdgcn_s_wait_tensorcnt((short)1);
      else        __builtin_amdgcn_s_wait_tensorcnt((short)0);
    }
    __syncthreads();

    FragBF fa, fb[4];
    const __bf16* arow = As + (wv * 16 + nl) * LDA + ks * BKt;
    fa.u[0] = *(const uint4*)(arow + 16 * g);
    fa.u[1] = *(const uint4*)(arow + 16 * g + 8);
    const __bf16* bbuf = &Bs[s & 1][0];
    #pragma unroll
    for (int ss = 0; ss < 4; ++ss) {
      const __bf16* brow = bbuf + (ss * 16 + nl) * LDB;
      fb[ss].u[0] = *(const uint4*)(brow + 16 * g);
      fb[ss].u[1] = *(const uint4*)(brow + 16 * g + 8);
    }
    #pragma unroll
    for (int ss = 0; ss < 4; ++ss)
      accR[ss] = __builtin_amdgcn_wmma_f32_16x16x32_bf16(
          false, fa.v, false, fb[ss].v, (short)0, accR[ss], false, false);
    __syncthreads();
  }

  // score[b,h,t,e] = score_row + colmax ; idxcol[b,h,t] = argmax col
  const size_t sobase = (size_t)bh * TT * BLK;
  #pragma unroll
  for (int s = 0; s < 4; ++s) {
    #pragma unroll
    for (int r = 0; r < 8; ++r) {
      int ml = wv * 16 + r + 8 * g;
      int t  = m0 + ml;
      int e  = s * 16 + nl;
      scoreOut[sobase + (size_t)t * BLK + e] = accR[s][r] + rmax[r];
    }
  }
  if (nl == 0) {
    #pragma unroll
    for (int r = 0; r < 8; ++r) {
      int ml = wv * 16 + r + 8 * g;
      idxOut[bh * TT + m0 + ml] = rarg[r];
    }
  }
}

// ---------------------------------------------------------------------------
// Kernel 4: per (b,h,e2) row of the [B,H,64,2048] raw view: bitonic sort of
// 2048 (value,pos) keys in LDS, emit sorted top-1024 + dispatch + indices.
// ---------------------------------------------------------------------------
__global__ __launch_bounds__(256) void topk_kernel(
    const float* __restrict__ score, const int* __restrict__ idxcol,
    float* __restrict__ oScore, int* __restrict__ oIdx, int* __restrict__ oDisp)
{
  __shared__ unsigned long long keys[2048];
  const int tid = threadIdx.x;
  const int bh  = blockIdx.x >> 6;
  const int e2  = blockIdx.x & 63;
  const float* srow = score + (size_t)bh * TT * BLK + (size_t)e2 * 2048;

  // ascending-sort key = (~ord(value) << 32) | pos  -> descending value,
  // ties broken by smallest position (matches top_k)
  for (int p = tid; p < 2048; p += 256) {
    unsigned u = __float_as_uint(srow[p]);
    u = (u & 0x80000000u) ? ~u : (u ^ 0x80000000u);
    keys[p] = ((unsigned long long)(~u) << 32) | (unsigned)p;
  }
  __syncthreads();

  for (int k = 2; k <= 2048; k <<= 1) {
    for (int j = k >> 1; j > 0; j >>= 1) {
      for (int i = tid; i < 2048; i += 256) {
        int ixj = i ^ j;
        if (ixj > i) {
          bool up = ((i & k) == 0);
          unsigned long long a = keys[i], b = keys[ixj];
          if ((a > b) == up) { keys[i] = b; keys[ixj] = a; }
        }
      }
      __syncthreads();
    }
  }

  const size_t base = ((size_t)bh * BLK + e2) * 1024;
  for (int jo = tid; jo < 1024; jo += 256) {
    int p = (int)(keys[jo] & 0xffffffffu);
    oScore[base + jo] = srow[p];
    oDisp[base + jo]  = p;
    int f  = e2 * 2048 + p;            // flat index in the raw [T,64] buffer
    int tp = f >> 6;                   // original t
    int ep = f & 63;                   // original e (row-key index)
    oIdx[base + jo] = ep * COLK + idxcol[bh * TT + tp];
  }
}

// ---------------------------------------------------------------------------
extern "C" void kernel_launch(void* const* d_in, const int* in_sizes, int n_in,
                              void* d_out, int out_size, void* d_ws, size_t ws_size,
                              hipStream_t stream)
{
  const float* x      = (const float*)d_in[0];
  const float* conv_w = (const float*)d_in[1];
  const float* conv_b = (const float*)d_in[2];
  const float* lin_w  = (const float*)d_in[3];
  const float* lin_b  = (const float*)d_in[4];
  const float* rowk   = (const float*)d_in[5];
  const float* colk   = (const float*)d_in[6];
  (void)in_sizes; (void)n_in; (void)out_size; (void)ws_size;

  char* ws = (char*)d_ws;
  __bf16* Y  = (__bf16*)(ws);               // 16 MB  [B*T, C]
  __bf16* W  = (__bf16*)(ws + 16777216);    //  2 MB  [QD, C]
  __bf16* Q  = (__bf16*)(ws + 18874368);    // 16 MB  [B*T, QD] (flat == view)
  __bf16* CK = (__bf16*)(ws + 35651584);    //  2 MB  [COL, H, D]
  __bf16* RK = (__bf16*)(ws + 37748736);    // 128 KB [64, H, D]
  float*  SC = (float*)(ws + 37879808);     //  8 MB  [B,H,T,64]
  int*    IX = (int*)(ws + 46268416);       // 128 KB [B,H,T]

  conv_pack_kernel<<<(BB * TT * CC) / 256, 256, 0, stream>>>(x, conv_w, conv_b, Y);
  cvt_bf16_kernel<<<(QD * CC) / 256, 256, 0, stream>>>(lin_w, W, QD * CC);
  cvt_bf16_kernel<<<(COLK * HH * DD) / 256, 256, 0, stream>>>(colk, CK, COLK * HH * DD);
  cvt_bf16_kernel<<<(BLK * HH * DD) / 256, 256, 0, stream>>>(rowk, RK, BLK * HH * DD);

  gemm_q_wmma<<<dim3(QD / 128, (BB * TT) / 128), 256, 0, stream>>>(Y, W, lin_b, Q);
  scores_wmma<<<dim3(TT / 64, BB * HH), 128, 0, stream>>>(Q, CK, RK, SC, IX);

  float* oS = (float*)d_out;
  int*   oI = (int*)((char*)d_out + (size_t)4 * 1048576);
  int*   oD = (int*)((char*)d_out + (size_t)8 * 1048576);
  topk_kernel<<<BB * HH * BLK, 256, 0, stream>>>(SC, IX, oS, oI, oD);
}
